// Mask2FormerDecoder_2267742732944
// MI455X (gfx1250) — matrix operand order repaired and checked by reading, not compiled, not run on hardware
//
#include <hip/hip_runtime.h>
#include <hip/hip_bf16.h>

// ---------------- types ----------------
typedef _Float16 half8  __attribute__((ext_vector_type(8)));
typedef _Float16 half16 __attribute__((ext_vector_type(16)));
typedef float    fl8    __attribute__((ext_vector_type(8)));

#define CAT16(a,b) __builtin_shufflevector(a,b,0,1,2,3,4,5,6,7,8,9,10,11,12,13,14,15)

// ---------------- problem constants ----------------
constexpr int Bn  = 8;     // batch
constexpr int Cc  = 256;   // d_model
constexpr int Ss  = 4096;  // H*W
constexpr int Nq  = 100;   // queries
constexpr int NqP = 112;   // queries padded to 16
constexpr int NH  = 8;     // heads
constexpr int Hd  = 32;    // head dim
constexpr int Fn  = 1024;  // ffn hidden
constexpr int Lx  = 6;     // layers
constexpr int NC  = 20;    // classes+1
constexpr int SPs = 128;   // self-attn padded key length (mult of 64)

enum : int { E_P32 = 0, E_CLASS, E_RESID, E_RELU, E_F16, E_QKVS, E_QC, E_KVC };

__device__ __forceinline__ fl8 wmma_f16(half16 a, half16 b, fl8 c) {
    return __builtin_amdgcn_wmma_f32_16x16x32_f16(false, a, false, b, (short)0, c, false, false);
}

// ---------------- generic WMMA GEMM: D = A(MxK) @ W(NxK)^T (+epilogue) ----------------
// Wave tile: 16(M) x 64(N)  (A fragment reused across 4 N-subtiles)
// Block tile: 16(M) x 256(N) with 4 waves.   All N must be multiples of 64.
struct GP {
    const _Float16* A; long long sA; int lda;   // per-batch stride / leading dim (halfs)
    const _Float16* W; long long sW; int ldw;
    const float*    bias;                        // may be null (E_P32)
    const float*    R;  long long sR;            // residual fp32 (E_RESID)
    float*          O32; long long sO32;
    _Float16*       O16; long long sO16;
    _Float16*       Qh; _Float16* Kh; _Float16* Vt;   // head-scatter outs
    int M, N, K, Mvalid, ldo, Skv;
};

template<int EPI>
__global__ __launch_bounds__(128) void gemm_k(GP p) {
    const int lane   = threadIdx.x & 31;
    const int wv     = threadIdx.x >> 5;
    const int b      = blockIdx.z;
    const int mBase  = blockIdx.x * 16;
    const int nBase0 = blockIdx.y * 256 + wv * 64;
    if (nBase0 >= p.N) return;
    const int l15 = lane & 15, lhi = lane >> 4;
    // A fragment: row = mBase+l15; halves [k0+8*lhi .. +7] and [+16 .. +23]   (ISA 16-bit A 16x32)
    const _Float16* Ap = p.A + (long long)b * p.sA + (long long)(mBase + l15) * p.lda + lhi * 8;
    // B fragment from W(N,K): row = n+l15; halves [k0+8*lhi .. +7] and [+16 .. +23] (same as A)
    const _Float16* Wp = p.W + (long long)b * p.sW + (long long)(nBase0 + l15) * p.ldw + lhi * 8;
    const long long wstep = (long long)16 * p.ldw;
    fl8 acc[4] = {{}, {}, {}, {}};
    for (int k0 = 0; k0 < p.K; k0 += 32) {
        half8 a0 = *(const half8*)(Ap + k0);
        half8 a1 = *(const half8*)(Ap + k0 + 16);
        half16 af = CAT16(a0, a1);
#pragma unroll
        for (int j = 0; j < 4; ++j) {
            const _Float16* wp = Wp + j * wstep + k0;
            half16 wf = CAT16(*(const half8*)wp, *(const half8*)(wp + 16));
            acc[j] = wmma_f16(af, wf, acc[j]);
        }
    }
    const int mTop = mBase + lhi * 8;
#pragma unroll
    for (int j = 0; j < 4; ++j) {
        const int n = nBase0 + j * 16 + l15;
#pragma unroll
        for (int v = 0; v < 8; ++v) {
            const int m = mTop + v;             // D layout: M = v + 8*(lane>>4), N = lane&15
            float val = acc[j][v];
            const bool live = (m < p.Mvalid);
            if (EPI == E_P32) {
                p.O32[(long long)b * p.sO32 + (long long)m * p.ldo + n] = live ? val : 0.f;
            } else if (EPI == E_CLASS) {
                if (live && n < NC) p.O32[((long long)b * Nq + m) * NC + n] = val + p.bias[n];
            } else if (EPI == E_RESID) {
                float o = live ? (val + p.bias[n] + p.R[(long long)b * p.sR + (long long)m * p.ldo + n]) : 0.f;
                p.O32[(long long)b * p.sO32 + (long long)m * p.ldo + n] = o;
            } else if (EPI == E_RELU) {
                float o = val + p.bias[n]; o = o > 0.f ? o : 0.f;
                p.O16[(long long)b * p.sO16 + (long long)m * p.ldo + n] = live ? (_Float16)o : (_Float16)0.f;
            } else if (EPI == E_F16) {
                float o = val + p.bias[n];
                p.O16[(long long)b * p.sO16 + (long long)m * p.ldo + n] = live ? (_Float16)o : (_Float16)0.f;
            } else if (EPI == E_QKVS) {
                float o = val + p.bias[n];
                _Float16 hv = live ? (_Float16)o : (_Float16)0.f;
                int seg = n >> 8, hh = (n >> 5) & 7, dd = n & 31;
                if (seg == 0)      p.Qh[((long long)(b * NH + hh) * NqP + m) * Hd + dd] = hv;
                else if (seg == 1) p.Kh[((long long)(b * NH + hh) * p.Skv + m) * Hd + dd] = hv;
                else               p.Vt[((long long)(b * NH + hh) * Hd + dd) * p.Skv + m] = hv;
            } else if (EPI == E_QC) {
                float o = val + p.bias[n];
                _Float16 hv = live ? (_Float16)o : (_Float16)0.f;
                int hh = n >> 5, dd = n & 31;
                p.Qh[((long long)(b * NH + hh) * NqP + m) * Hd + dd] = hv;
            } else if (EPI == E_KVC) {
                _Float16 hv = (_Float16)(val + p.bias[n]);       // all S rows valid
                int seg = n >> 8, hh = (n >> 5) & 7, dd = n & 31;
                if (seg == 0) p.Kh[((long long)(b * NH + hh) * p.Skv + m) * Hd + dd] = hv;
                else          p.Vt[((long long)(b * NH + hh) * Hd + dd) * p.Skv + m] = hv;
            }
        }
    }
}

// ---------------- flash attention (16 queries/wave, 64 keys/iter, online softmax) ----------------
struct FP {
    const _Float16* Q;   // (B,NH,NqP,Hd)
    const _Float16* K;   // (B,NH,SP,Hd)
    const _Float16* Vt;  // (B,NH,Hd,SP)
    const float*    mask; // pred_mask fp32 (B,NqP,Ss), or null
    _Float16*       Out;  // (B,NqP,C) head-concatenated
    int SP, keyLimit;
    float scale;
};

__global__ __launch_bounds__(32) void flash_k(FP p) {
    __shared__ alignas(16) _Float16 lds[16 * 64];
    const int lane = threadIdx.x;
    const int b = blockIdx.z, hh = blockIdx.y, mBase = blockIdx.x * 16;
    const int l15 = lane & 15, lhi = lane >> 4;
    const int kba = lhi * 8, kbb = lhi * 8;
    const _Float16* Qb = p.Q  + (long long)(b * NH + hh) * NqP * Hd;
    const _Float16* Kb = p.K  + (long long)(b * NH + hh) * p.SP * Hd;
    const _Float16* Vb = p.Vt + (long long)(b * NH + hh) * Hd * p.SP;
    const _Float16* qp = Qb + (mBase + l15) * Hd + kba;
    const half16 qa = CAT16(*(const half8*)qp, *(const half8*)(qp + 16));
    fl8 o0 = {}, o1 = {};
    float rmax[8], rsum[8];
#pragma unroll
    for (int v = 0; v < 8; ++v) { rmax[v] = -3.0e38f; rsum[v] = 0.f; }
    const int mTop = mBase + lhi * 8;
    for (int kc = 0; kc < p.SP; kc += 64) {
        // ---- scores: 4 x (16q x 16k) WMMAs ----
        fl8 s[4];
#pragma unroll
        for (int j = 0; j < 4; ++j) {
            const _Float16* kp = Kb + (long long)(kc + j * 16 + l15) * Hd + kbb;
            half16 kf = CAT16(*(const half8*)kp, *(const half8*)(kp + 16));
            fl8 z = {};
            s[j] = wmma_f16(qa, kf, z);
        }
        // ---- online softmax over 64 keys ----
        float e[4][8];
#pragma unroll
        for (int v = 0; v < 8; ++v) {
            const int m = mTop + v;
            float f[4];
#pragma unroll
            for (int j = 0; j < 4; ++j) {
                const int key = kc + j * 16 + l15;
                float fv = s[j][v] * p.scale;
                if (p.mask) {
                    const float* mr = p.mask + ((long long)b * NqP + m) * Ss;
                    fv += (mr[key] < 0.5f) ? -1.0e9f : 0.f;
                }
                if (key >= p.keyLimit) fv = -3.0e38f;
                f[j] = fv;
            }
            float cm = fmaxf(fmaxf(f[0], f[1]), fmaxf(f[2], f[3]));
            cm = fmaxf(cm, __shfl_xor(cm, 1, 32));
            cm = fmaxf(cm, __shfl_xor(cm, 2, 32));
            cm = fmaxf(cm, __shfl_xor(cm, 4, 32));
            cm = fmaxf(cm, __shfl_xor(cm, 8, 32));
            float nm = fmaxf(rmax[v], cm);
            float corr = expf(rmax[v] - nm);
            float rs = 0.f;
#pragma unroll
            for (int j = 0; j < 4; ++j) { e[j][v] = expf(f[j] - nm); rs += e[j][v]; }
            rs += __shfl_xor(rs, 1, 32); rs += __shfl_xor(rs, 2, 32);
            rs += __shfl_xor(rs, 4, 32); rs += __shfl_xor(rs, 8, 32);
            rsum[v] = rsum[v] * corr + rs;
            o0[v] *= corr; o1[v] *= corr;
            rmax[v] = nm;
        }
        // ---- P tile: D-layout -> A-layout via LDS (wave-local) ----
#pragma unroll
        for (int v = 0; v < 8; ++v) {
            const int r = lhi * 8 + v;
#pragma unroll
            for (int j = 0; j < 4; ++j)
                lds[r * 64 + j * 16 + l15] = (_Float16)e[j][v];
        }
        __syncthreads();
        const _Float16* lp = lds + l15 * 64 + kba;
        half16 pa0 = CAT16(*(const half8*)(lp),      *(const half8*)(lp + 16));
        half16 pa1 = CAT16(*(const half8*)(lp + 32), *(const half8*)(lp + 48));
        // ---- out += P @ V (Vt rows = head dim, cols = keys) ----
        const _Float16* vpA = Vb + (long long)l15 * p.SP + kc + kbb;
        const _Float16* vpB = vpA + (long long)16 * p.SP;
        half16 v00 = CAT16(*(const half8*)vpA,        *(const half8*)(vpA + 16));
        half16 v01 = CAT16(*(const half8*)vpB,        *(const half8*)(vpB + 16));
        half16 v10 = CAT16(*(const half8*)(vpA + 32), *(const half8*)(vpA + 48));
        half16 v11 = CAT16(*(const half8*)(vpB + 32), *(const half8*)(vpB + 48));
        o0 = wmma_f16(pa0, v00, o0);
        o1 = wmma_f16(pa0, v01, o1);
        o0 = wmma_f16(pa1, v10, o0);
        o1 = wmma_f16(pa1, v11, o1);
        __syncthreads();
    }
#pragma unroll
    for (int v = 0; v < 8; ++v) {
        const int m = mTop + v;
        const float inv = 1.f / rsum[v];
        const bool live = (m < Nq);
        _Float16* orow = p.Out + ((long long)b * NqP + m) * Cc + hh * Hd;
        orow[l15]      = live ? (_Float16)(o0[v] * inv) : (_Float16)0.f;
        orow[16 + l15] = live ? (_Float16)(o1[v] * inv) : (_Float16)0.f;
    }
}

// ---------------- small helper kernels ----------------
__global__ void cvt_k(const float* in, _Float16* out, long long n) {
    long long i = (long long)blockIdx.x * blockDim.x + threadIdx.x;
    if (i < n) out[i] = (_Float16)in[i];
}

__global__ __launch_bounds__(256) void transp_k(const float* in, _Float16* out) {
    // (B,C,S) fp32 -> (B,S,C) fp16
    __shared__ float t[16][17];
    int b = blockIdx.z, s0 = blockIdx.x * 16, c0 = blockIdx.y * 16;
    int tx = threadIdx.x, ty = threadIdx.y;
    t[ty][tx] = in[((long long)b * Cc + c0 + ty) * Ss + s0 + tx];
    __syncthreads();
    out[((long long)b * Ss + s0 + ty) * Cc + c0 + tx] = (_Float16)t[tx][ty];
}

__global__ void qinit_k(const float* q, float* q32, _Float16* q16) {
    long long i = (long long)blockIdx.x * blockDim.x + threadIdx.x;
    if (i >= (long long)Bn * NqP * Cc) return;
    int c = (int)(i & (Cc - 1));
    int m = (int)((i >> 8) % NqP);
    float v = (m < Nq) ? q[m * Cc + c] : 0.f;
    q32[i] = v; q16[i] = (_Float16)v;
}

__global__ __launch_bounds__(128) void ln_k(const float* x, const float* g, const float* bt,
                                            float* y32, _Float16* y16) {
    int lane = threadIdx.x & 31;
    int row = blockIdx.x * 4 + (threadIdx.x >> 5);           // row < Bn*NqP
    const float* xr = x + (long long)row * Cc;
    float v[8]; float s = 0.f, ss = 0.f;
#pragma unroll
    for (int i = 0; i < 8; ++i) { v[i] = xr[lane * 8 + i]; s += v[i]; ss += v[i] * v[i]; }
    for (int msk = 1; msk < 32; msk <<= 1) { s += __shfl_xor(s, msk, 32); ss += __shfl_xor(ss, msk, 32); }
    float mu = s * (1.f / Cc);
    float var = ss * (1.f / Cc) - mu * mu;
    float r = rsqrtf(var + 1e-5f);
    int m = row % NqP;
    float* yr = y32 + (long long)row * Cc;
    _Float16* hr = y16 + (long long)row * Cc;
#pragma unroll
    for (int i = 0; i < 8; ++i) {
        int c = lane * 8 + i;
        float o = (m < Nq) ? ((v[i] - mu) * r * g[c] + bt[c]) : 0.f;
        yr[c] = o; hr[c] = (_Float16)o;
    }
}

__global__ void predcopy_k(const float* pred, float* out) {
    long long i = (long long)blockIdx.x * blockDim.x + threadIdx.x;
    if (i >= (long long)Bn * Nq * Ss) return;
    int s = (int)(i & (Ss - 1));
    long long r = i >> 12;
    int m = (int)(r % Nq), b = (int)(r / Nq);
    out[i] = pred[((long long)b * NqP + m) * Ss + s];
}

// ---------------- host orchestration ----------------
extern "C" void kernel_launch(void* const* d_in, const int* in_sizes, int n_in,
                              void* d_out, int out_size, void* d_ws, size_t ws_size,
                              hipStream_t stream) {
    (void)in_sizes; (void)n_in; (void)out_size; (void)ws_size;
    const float* memory   = (const float*)d_in[0];
    const float* pixf     = (const float*)d_in[1];
    const float* queries  = (const float*)d_in[2];
    const float* sa_in_w  = (const float*)d_in[3];
    const float* sa_in_b  = (const float*)d_in[4];
    const float* sa_out_w = (const float*)d_in[5];
    const float* sa_out_b = (const float*)d_in[6];
    const float* ca_in_w  = (const float*)d_in[7];
    const float* ca_in_b  = (const float*)d_in[8];
    const float* ca_out_w = (const float*)d_in[9];
    const float* ca_out_b = (const float*)d_in[10];
    const float* ffn_w1   = (const float*)d_in[11];
    const float* ffn_b1   = (const float*)d_in[12];
    const float* ffn_w2   = (const float*)d_in[13];
    const float* ffn_b2   = (const float*)d_in[14];
    const float* n1g = (const float*)d_in[15];
    const float* n1b = (const float*)d_in[16];
    const float* n2g = (const float*)d_in[17];
    const float* n2b = (const float*)d_in[18];
    const float* n3g = (const float*)d_in[19];
    const float* n3b = (const float*)d_in[20];
    const float* cls_w = (const float*)d_in[21];
    const float* cls_b = (const float*)d_in[22];
    const float* mh_w1 = (const float*)d_in[23];
    const float* mh_b1 = (const float*)d_in[24];
    const float* mh_w2 = (const float*)d_in[25];
    const float* mh_b2 = (const float*)d_in[26];

    char* ws = (char*)d_ws;
    size_t off = 0;
    auto alloc = [&](size_t bytes) -> char* {
        char* p = ws + off;
        off = (off + bytes + 255) & ~(size_t)255;
        return p;
    };
    // fp16 weights
    _Float16* sa_in16  = (_Float16*)alloc((size_t)Lx * 3 * Cc * Cc * 2);
    _Float16* sa_out16 = (_Float16*)alloc((size_t)Lx * Cc * Cc * 2);
    _Float16* ca_in16  = (_Float16*)alloc((size_t)Lx * 3 * Cc * Cc * 2);
    _Float16* ca_out16 = (_Float16*)alloc((size_t)Lx * Cc * Cc * 2);
    _Float16* f1_16    = (_Float16*)alloc((size_t)Lx * Fn * Cc * 2);
    _Float16* f2_16    = (_Float16*)alloc((size_t)Lx * Cc * Fn * 2);
    _Float16* mh1_16   = (_Float16*)alloc((size_t)Cc * Cc * 2);
    _Float16* mh2_16   = (_Float16*)alloc((size_t)Cc * Cc * 2);
    _Float16* cls16    = (_Float16*)alloc((size_t)64 * Cc * 2);       // 20 rows valid, padded to 64
    // fp16 tensors
    _Float16* memT16 = (_Float16*)alloc((size_t)Bn * Ss * Cc * 2);
    _Float16* pfT16  = (_Float16*)alloc((size_t)Bn * Ss * Cc * 2);
    _Float16* q16    = (_Float16*)alloc((size_t)Bn * NqP * Cc * 2);
    _Float16* ao16   = (_Float16*)alloc((size_t)Bn * NqP * Cc * 2);
    _Float16* hid16  = (_Float16*)alloc((size_t)Bn * NqP * Fn * 2);
    _Float16* t16    = (_Float16*)alloc((size_t)Bn * NqP * Cc * 2);
    _Float16* me16   = (_Float16*)alloc((size_t)Bn * NqP * Cc * 2);
    _Float16* Qs     = (_Float16*)alloc((size_t)Bn * NH * NqP * Hd * 2);
    _Float16* Ks     = (_Float16*)alloc((size_t)Bn * NH * SPs * Hd * 2);
    _Float16* Vts    = (_Float16*)alloc((size_t)Bn * NH * Hd * SPs * 2);
    _Float16* Qc     = (_Float16*)alloc((size_t)Bn * NH * NqP * Hd * 2);
    _Float16* Kc     = (_Float16*)alloc((size_t)Bn * NH * Ss * Hd * 2);
    _Float16* Vtc    = (_Float16*)alloc((size_t)Bn * NH * Hd * Ss * 2);
    // fp32 tensors
    float* q32    = (float*)alloc((size_t)Bn * NqP * Cc * 4);
    float* tmp32  = (float*)alloc((size_t)Bn * NqP * Cc * 4);
    float* pred32 = (float*)alloc((size_t)Bn * NqP * Ss * 4);

    auto cvt = [&](const float* src, _Float16* dst, long long n) {
        cvt_k<<<(unsigned)((n + 255) / 256), 256, 0, stream>>>(src, dst, n);
    };
    cvt(sa_in_w,  sa_in16,  (long long)Lx * 3 * Cc * Cc);
    cvt(sa_out_w, sa_out16, (long long)Lx * Cc * Cc);
    cvt(ca_in_w,  ca_in16,  (long long)Lx * 3 * Cc * Cc);
    cvt(ca_out_w, ca_out16, (long long)Lx * Cc * Cc);
    cvt(ffn_w1,   f1_16,    (long long)Lx * Fn * Cc);
    cvt(ffn_w2,   f2_16,    (long long)Lx * Cc * Fn);
    cvt(mh_w1,    mh1_16,   (long long)Cc * Cc);
    cvt(mh_w2,    mh2_16,   (long long)Cc * Cc);
    cvt(cls_w,    cls16,    (long long)NC * Cc);

    transp_k<<<dim3(Ss / 16, Cc / 16, Bn), dim3(16, 16), 0, stream>>>(memory, memT16);
    transp_k<<<dim3(Ss / 16, Cc / 16, Bn), dim3(16, 16), 0, stream>>>(pixf, pfT16);
    qinit_k<<<(Bn * NqP * Cc) / 256, 256, 0, stream>>>(queries, q32, q16);

    const float scale = 0.17677669529663689f;   // 1/sqrt(32)
    const int lnGrid = (Bn * NqP) / 4;
    auto gy = [](int N) { return (unsigned)((N + 255) / 256); };

    for (int l = 0; l < Lx; ++l) {
        // ---- self attention: QKV projection ----
        { GP g{}; g.A = q16; g.sA = NqP * Cc; g.lda = Cc;
          g.W = sa_in16 + (size_t)l * 3 * Cc * Cc; g.sW = 0; g.ldw = Cc;
          g.bias = sa_in_b + l * 3 * Cc;
          g.M = NqP; g.N = 3 * Cc; g.K = Cc; g.Mvalid = Nq; g.Skv = SPs;
          g.Qh = Qs; g.Kh = Ks; g.Vt = Vts;
          gemm_k<E_QKVS><<<dim3(NqP / 16, gy(3 * Cc), Bn), 128, 0, stream>>>(g); }
        // ---- self attention core ----
        { FP f{Qs, Ks, Vts, nullptr, ao16, SPs, Nq, scale};
          flash_k<<<dim3(NqP / 16, NH, Bn), 32, 0, stream>>>(f); }
        // ---- out proj + residual, then norm1 ----
        { GP g{}; g.A = ao16; g.sA = NqP * Cc; g.lda = Cc;
          g.W = sa_out16 + (size_t)l * Cc * Cc; g.sW = 0; g.ldw = Cc;
          g.bias = sa_out_b + l * Cc; g.R = q32; g.sR = NqP * Cc;
          g.O32 = tmp32; g.sO32 = NqP * Cc; g.ldo = Cc;
          g.M = NqP; g.N = Cc; g.K = Cc; g.Mvalid = Nq;
          gemm_k<E_RESID><<<dim3(NqP / 16, gy(Cc), Bn), 128, 0, stream>>>(g); }
        ln_k<<<lnGrid, 128, 0, stream>>>(tmp32, n1g + l * Cc, n1b + l * Cc, q32, q16);

        // ---- cross attention: Q projection ----
        { GP g{}; g.A = q16; g.sA = NqP * Cc; g.lda = Cc;
          g.W = ca_in16 + (size_t)l * 3 * Cc * Cc; g.sW = 0; g.ldw = Cc;
          g.bias = ca_in_b + l * 3 * Cc;
          g.M = NqP; g.N = Cc; g.K = Cc; g.Mvalid = Nq; g.Qh = Qc;
          gemm_k<E_QC><<<dim3(NqP / 16, gy(Cc), Bn), 128, 0, stream>>>(g); }
        // ---- cross attention: K,V projections over memory ----
        { GP g{}; g.A = memT16; g.sA = (long long)Ss * Cc; g.lda = Cc;
          g.W = ca_in16 + (size_t)l * 3 * Cc * Cc + (size_t)Cc * Cc; g.sW = 0; g.ldw = Cc;
          g.bias = ca_in_b + l * 3 * Cc + Cc;
          g.M = Ss; g.N = 2 * Cc; g.K = Cc; g.Mvalid = Ss; g.Skv = Ss;
          g.Kh = Kc; g.Vt = Vtc;
          gemm_k<E_KVC><<<dim3(Ss / 16, gy(2 * Cc), Bn), 128, 0, stream>>>(g); }
        // ---- cross attention core (masked from layer 1 on) ----
        { FP f{Qc, Kc, Vtc, (l > 0 ? pred32 : nullptr), ao16, Ss, Ss, scale};
          flash_k<<<dim3(NqP / 16, NH, Bn), 32, 0, stream>>>(f); }
        // ---- out proj + residual, then norm2 ----
        { GP g{}; g.A = ao16; g.sA = NqP * Cc; g.lda = Cc;
          g.W = ca_out16 + (size_t)l * Cc * Cc; g.sW = 0; g.ldw = Cc;
          g.bias = ca_out_b + l * Cc; g.R = q32; g.sR = NqP * Cc;
          g.O32 = tmp32; g.sO32 = NqP * Cc; g.ldo = Cc;
          g.M = NqP; g.N = Cc; g.K = Cc; g.Mvalid = Nq;
          gemm_k<E_RESID><<<dim3(NqP / 16, gy(Cc), Bn), 128, 0, stream>>>(g); }
        ln_k<<<lnGrid, 128, 0, stream>>>(tmp32, n2g + l * Cc, n2b + l * Cc, q32, q16);

        // ---- FFN ----
        { GP g{}; g.A = q16; g.sA = NqP * Cc; g.lda = Cc;
          g.W = f1_16 + (size_t)l * Fn * Cc; g.sW = 0; g.ldw = Cc;
          g.bias = ffn_b1 + l * Fn;
          g.O16 = hid16; g.sO16 = (long long)NqP * Fn; g.ldo = Fn;
          g.M = NqP; g.N = Fn; g.K = Cc; g.Mvalid = Nq;
          gemm_k<E_RELU><<<dim3(NqP / 16, gy(Fn), Bn), 128, 0, stream>>>(g); }
        { GP g{}; g.A = hid16; g.sA = (long long)NqP * Fn; g.lda = Fn;
          g.W = f2_16 + (size_t)l * Cc * Fn; g.sW = 0; g.ldw = Fn;
          g.bias = ffn_b2 + l * Cc; g.R = q32; g.sR = NqP * Cc;
          g.O32 = tmp32; g.sO32 = NqP * Cc; g.ldo = Cc;
          g.M = NqP; g.N = Cc; g.K = Fn; g.Mvalid = Nq;
          gemm_k<E_RESID><<<dim3(NqP / 16, gy(Cc), Bn), 128, 0, stream>>>(g); }
        ln_k<<<lnGrid, 128, 0, stream>>>(tmp32, n3g + l * Cc, n3b + l * Cc, q32, q16);

        // ---- mask head: me = relu(q@mh1^T+b1)@mh2^T+b2 ; pred = me @ pf ----
        { GP g{}; g.A = q16; g.sA = NqP * Cc; g.lda = Cc;
          g.W = mh1_16; g.sW = 0; g.ldw = Cc; g.bias = mh_b1;
          g.O16 = t16; g.sO16 = NqP * Cc; g.ldo = Cc;
          g.M = NqP; g.N = Cc; g.K = Cc; g.Mvalid = Nq;
          gemm_k<E_RELU><<<dim3(NqP / 16, gy(Cc), Bn), 128, 0, stream>>>(g); }
        { GP g{}; g.A = t16; g.sA = NqP * Cc; g.lda = Cc;
          g.W = mh2_16; g.sW = 0; g.ldw = Cc; g.bias = mh_b2;
          g.O16 = me16; g.sO16 = NqP * Cc; g.ldo = Cc;
          g.M = NqP; g.N = Cc; g.K = Cc; g.Mvalid = Nq;
          gemm_k<E_F16><<<dim3(NqP / 16, gy(Cc), Bn), 128, 0, stream>>>(g); }
        { GP g{}; g.A = me16; g.sA = NqP * Cc; g.lda = Cc;
          g.W = pfT16; g.sW = (long long)Ss * Cc; g.ldw = Cc; g.bias = nullptr;
          g.O32 = pred32; g.sO32 = (long long)NqP * Ss; g.ldo = Ss;
          g.M = NqP; g.N = Ss; g.K = Cc; g.Mvalid = Nq;
          gemm_k<E_P32><<<dim3(NqP / 16, gy(Ss), Bn), 128, 0, stream>>>(g); }
    }

    // ---- class logits -> d_out[0 .. B*Nq*NC)   (N padded to 64, stores guarded n<NC) ----
    { GP g{}; g.A = q16; g.sA = NqP * Cc; g.lda = Cc;
      g.W = cls16; g.sW = 0; g.ldw = Cc; g.bias = cls_b;
      g.O32 = (float*)d_out;
      g.M = NqP; g.N = 64; g.K = Cc; g.Mvalid = Nq;
      gemm_k<E_CLASS><<<dim3(NqP / 16, 1, Bn), 128, 0, stream>>>(g); }
    // ---- pred_mask (layer 5) -> d_out[B*Nq*NC ..) compacted to (B,100,4096) ----
    predcopy_k<<<(Bn * Nq * Ss + 255) / 256, 256, 0, stream>>>(
        pred32, (float*)d_out + (size_t)Bn * Nq * NC);
}
